// VisionSdpaAttention_15642270892544
// MI455X (gfx1250) — matrix operand order repaired and checked by reading, not compiled, not run on hardware
//
#include <hip/hip_runtime.h>
#include <math.h>

typedef __attribute__((ext_vector_type(16))) _Float16 v16h;
typedef __attribute__((ext_vector_type(8)))  _Float16 v8h;
typedef __attribute__((ext_vector_type(2)))  _Float16 v2h;
typedef __attribute__((ext_vector_type(8)))  float    v8f;

#define NUM_HEADS 16
#define HEAD_DIM  80
#define TOK_STRIDE (NUM_HEADS * HEAD_DIM)   // 1280 floats per token
#define WAVES_PER_BLOCK 4
#define QT 2                                 // query 16-row tiles per wave
#define NEG_BIG (-3.0e38f)

#define K_STRIDE 88   // LDS K row stride in halves (176B: conflict-free, 16B-aligned)
#define V_STRIDE 40   // LDS V row stride in halves (80B:  conflict-free, 16B-aligned)

#define CAT16(lo, hi) __builtin_shufflevector(lo, hi, 0,1,2,3,4,5,6,7,8,9,10,11,12,13,14,15)

// ---- CDNA5 async global->LDS copy (16B per lane), tracked by ASYNCcnt ----
__device__ __forceinline__ void async_cp16(_Float16* dst_lds, const _Float16* src_glob) {
  unsigned l = (unsigned)(unsigned long long)dst_lds;       // low 32 bits = LDS offset
  unsigned long long g = (unsigned long long)src_glob;
  asm volatile("global_load_async_to_lds_b128 %0, %1, off"
               :: "v"(l), "v"(g) : "memory");
}
__device__ __forceinline__ void wait_async_all() {
  asm volatile("s_wait_asynccnt 0x0" ::: "memory");
}

// ---------------- pre-pass: fp32 -> f16 repack ----------------
// Qf/Kf layout: [head][token][d] (f16, Q pre-scaled by SCALE*log2e)
__global__ void cvt_qk_kernel(const float* __restrict__ Q, const float* __restrict__ K,
                              _Float16* __restrict__ Qf, _Float16* __restrict__ Kf,
                              int S, float qscale, long total_pairs) {
  long i = (long)blockIdx.x * blockDim.x + threadIdx.x;
  if (i >= total_pairs) return;
  const int PP = HEAD_DIM / 2;
  int dp  = (int)(i % PP);
  long ht = i / PP;
  int tok = (int)(ht % S);
  int h   = (int)(ht / S);
  size_t src = (size_t)tok * TOK_STRIDE + (size_t)h * HEAD_DIM + dp * 2;
  size_t dst = ((size_t)h * S + tok) * HEAD_DIM + dp * 2;
  v2h qo, ko;
  qo[0] = (_Float16)(Q[src] * qscale);
  qo[1] = (_Float16)(Q[src + 1] * qscale);
  ko[0] = (_Float16)K[src];
  ko[1] = (_Float16)K[src + 1];
  *(v2h*)(Qf + dst) = qo;
  *(v2h*)(Kf + dst) = ko;
}

// Vt layout: [head][d][kv] (f16), kv row stride Sv (8-aligned)
__global__ void cvt_vt_kernel(const float* __restrict__ V, _Float16* __restrict__ Vt,
                              int S, int Sv, long total_pairs) {
  long i = (long)blockIdx.x * blockDim.x + threadIdx.x;
  if (i >= total_pairs) return;
  const int S2 = S >> 1;
  int kvp = (int)(i % S2);
  long hd = i / S2;                        // h*80 + d
  int kv  = kvp * 2;
  v2h o;
  o[0] = (_Float16)V[(size_t)kv * TOK_STRIDE + hd];
  o[1] = (_Float16)V[(size_t)(kv + 1) * TOK_STRIDE + hd];
  *(v2h*)(Vt + hd * (size_t)Sv + kv) = o;
}

// ---------------- main flash-attention kernel (f16, LDS-staged K/V) ----------------
// Workgroup = 4 waves, same head, 128 consecutive query rows (same segment).
// K/V 32-token blocks staged into LDS once per workgroup via async copies,
// double-buffered; each wave computes its own QT query tiles from LDS.
__global__ __launch_bounds__(WAVES_PER_BLOCK * 32) void
fa_wmma_f16_kernel(const _Float16* __restrict__ Qf, const _Float16* __restrict__ Kf,
                   const _Float16* __restrict__ Vt, const int* __restrict__ cu,
                   int n_cu, float* __restrict__ Out, int S, int Sv) {
  const int tid  = threadIdx.x;
  const int lane = tid & 31;
  const int wid  = tid >> 5;
  const int lr   = lane & 15;
  const int half = lane >> 4;
  const int head = blockIdx.y;
  const int q_base_blk = blockIdx.x * (WAVES_PER_BLOCK * QT * 16);
  const int q_base     = q_base_blk + wid * (QT * 16);

  __shared__ __align__(16) _Float16 smemK[2][32 * K_STRIDE + 16];
  __shared__ __align__(16) _Float16 smemV[2][80 * V_STRIDE];

  const v16h vzero = {};

  // segment bounds: uniform across the whole block (cu boundaries 128-aligned)
  int seg_start = 0, seg_end = S;
  for (int i = 0; i + 1 < n_cu; ++i) {
    int a = cu[i], b = cu[i + 1];
    if (q_base_blk >= a && q_base_blk < b) { seg_start = a; seg_end = b; }
  }

  const _Float16* qhead  = Qf + (size_t)head * S * HEAD_DIM;
  const _Float16* khead  = Kf + (size_t)head * S * HEAD_DIM;
  const _Float16* vthead = Vt + (size_t)head * HEAD_DIM * (size_t)Sv;

  // ---- Q as B-operand: lane holds column q, k = 32c + 16*half + e
  v16h qb[QT][3];
#pragma unroll
  for (int qt = 0; qt < QT; ++qt) {
    int qrow = q_base + qt * 16 + lr;
    if (qrow >= S) qrow = S - 1;
    const _Float16* qp = qhead + (size_t)qrow * HEAD_DIM + half * 16;
#pragma unroll
    for (int c = 0; c < 3; ++c) {
      v8h lo = *(const v8h*)(qp + c * 32);
      v8h hi = *(const v8h*)(qp + c * 32 + 8);
      v16h q16 = CAT16(lo, hi);
      if (c == 2 && half == 1) q16 = vzero;   // d in [80,96) must be zero in B
      qb[qt][c] = q16;
    }
  }

  v8f  o[QT][5] = {};
  float m_run[QT], l_run[QT];
#pragma unroll
  for (int qt = 0; qt < QT; ++qt) { m_run[qt] = NEG_BIG; l_run[qt] = 0.0f; }

  // ---- cooperative staging of one 32-token K/V block into LDS
  auto stage = [&](int kvb, int b) {
    const _Float16* kg = khead + (size_t)kvb * HEAD_DIM;
    const _Float16* vg = vthead + kvb;
    // K: 32 rows x 11 chunks of 8 halves (cols 80..87 = global over-read pad)
#pragma unroll
    for (int it = 0; it < 3; ++it) {
      int cc = tid + it * (WAVES_PER_BLOCK * 32);
      if (cc < 32 * 11) {
        int row = cc / 11, col = cc % 11;
        async_cp16(&smemK[b][row * K_STRIDE + col * 8], kg + row * HEAD_DIM + col * 8);
      }
    }
    // V: 80 d-rows x 4 chunks of 8 halves
#pragma unroll
    for (int it = 0; it < 3; ++it) {
      int cc = tid + it * (WAVES_PER_BLOCK * 32);
      if (cc < 80 * 4) {
        int row = cc / 4, col = cc % 4;
        async_cp16(&smemV[b][row * V_STRIDE + col * 8], vg + (size_t)row * Sv + col * 8);
      }
    }
  };

  stage(seg_start, 0);
  wait_async_all();
  __syncthreads();

  int buf = 0;
  for (int kvb = seg_start; kvb < seg_end; kvb += 32, buf ^= 1) {
    const int rem = seg_end - kvb;
    if (kvb + 32 < seg_end) stage(kvb + 32, buf ^ 1);
    if (kvb + 64 < seg_end)
      __builtin_prefetch(khead + (size_t)(kvb + 64 + lane) * HEAD_DIM, 0, 3);

    // ---- S^T tiles from LDS K: st[qt][t] covers kv rows [kvb+16t, kvb+16t+16)
    v8f st[QT][2] = {};
#pragma unroll
    for (int t = 0; t < 2; ++t) {
      int lrow = t * 16 + lr;
      if (lrow > rem - 1) lrow = rem - 1;               // tail clamp (masked below)
      const _Float16* kl = &smemK[buf][lrow * K_STRIDE + half * 16];
#pragma unroll
      for (int c = 0; c < 3; ++c) {
        v8h lo = *(const v8h*)(kl + c * 32);
        v8h hi = *(const v8h*)(kl + c * 32 + 8);
        v16h ka = CAT16(lo, hi);                        // c==2 pad pairs with qb zeros
#pragma unroll
        for (int qt = 0; qt < QT; ++qt)
          st[qt][t] = __builtin_amdgcn_wmma_f32_16x16x32_f16(
              false, ka, false, qb[qt][c], (short)0, st[qt][t], false, false);
      }
    }

    // ---- online softmax + pack P^T as B-operand
    v16h pb[QT];
#pragma unroll
    for (int qt = 0; qt < QT; ++qt) {
      if (rem < 32) {
#pragma unroll
        for (int v = 0; v < 8; ++v) {
          if (v + 8 * half >= rem)      st[qt][0][v] = NEG_BIG;
          if (16 + v + 8 * half >= rem) st[qt][1][v] = NEG_BIG;
        }
      }
      float mblk = NEG_BIG;
#pragma unroll
      for (int v = 0; v < 8; ++v)
        mblk = fmaxf(mblk, fmaxf(st[qt][0][v], st[qt][1][v]));
      mblk = fmaxf(mblk, __shfl_xor(mblk, 16, 32));
      const float mnew = fmaxf(m_run[qt], mblk);
      if (__any(mblk > m_run[qt])) {                    // skip rescale when max unchanged
        const float corr = exp2f(m_run[qt] - mnew);
#pragma unroll
        for (int dt = 0; dt < 5; ++dt) o[qt][dt] *= corr;
        l_run[qt] *= corr;
      }
      m_run[qt] = mnew;

      float p0[8], p1[8], lblk = 0.0f;
#pragma unroll
      for (int v = 0; v < 8; ++v) {
        p0[v] = exp2f(st[qt][0][v] - mnew);
        p1[v] = exp2f(st[qt][1][v] - mnew);
        lblk += p0[v] + p1[v];
      }
      lblk += __shfl_xor(lblk, 16, 32);
      l_run[qt] += lblk;

#pragma unroll
      for (int v = 0; v < 8; ++v) {
        const float s0 = __shfl_xor(p0[v], 16, 32);
        const float s1 = __shfl_xor(p1[v], 16, 32);
        pb[qt][v]     = (_Float16)((half == 0) ? p0[v] : s1);
        pb[qt][8 + v] = (_Float16)((half == 0) ? s0    : p1[v]);
      }
    }

    // ---- O^T += V^T * P^T from LDS V (A lane row d = 16*dt + lr)
#pragma unroll
    for (int dt = 0; dt < 5; ++dt) {
      const _Float16* vl = &smemV[buf][(dt * 16 + lr) * V_STRIDE + half * 16];
      v8h lo = *(const v8h*)(vl);
      v8h hi = *(const v8h*)(vl + 8);
      v16h va = CAT16(lo, hi);
#pragma unroll
      for (int qt = 0; qt < QT; ++qt)
        o[qt][dt] = __builtin_amdgcn_wmma_f32_16x16x32_f16(
            false, va, false, pb[qt], (short)0, o[qt][dt], false, false);
    }

    wait_async_all();   // next buffer filled
    __syncthreads();    // all waves done reading current buffer
  }

  // ---- epilogue
#pragma unroll
  for (int qt = 0; qt < QT; ++qt) {
    const int qrow = q_base + qt * 16 + lr;
    if (qrow < S) {
      const float linv = 1.0f / l_run[qt];
      float* op = Out + (size_t)qrow * TOK_STRIDE + head * HEAD_DIM;
#pragma unroll
      for (int dt = 0; dt < 5; ++dt)
#pragma unroll
        for (int v = 0; v < 8; ++v)
          op[dt * 16 + v + 8 * half] = o[qt][dt][v] * linv;
    }
  }
}

// ---------------- fallback: direct fp32 kernel (round-1) ----------------
__global__ __launch_bounds__(WAVES_PER_BLOCK * 32) void
fa_wmma_f32_kernel(const float* __restrict__ Q, const float* __restrict__ K,
                   const float* __restrict__ V, const int* __restrict__ cu,
                   int n_cu, float* __restrict__ Out, int S, float scale_log2e) {
  const int lane = threadIdx.x & 31;
  const int wid  = threadIdx.x >> 5;
  const int lr   = lane & 15;
  const int half = lane >> 4;
  const int head   = blockIdx.y;
  const int q_base = (blockIdx.x * WAVES_PER_BLOCK + wid) * 16;
  if (q_base >= S) return;

  int seg_start = 0, seg_end = S;
  for (int i = 0; i + 1 < n_cu; ++i) {
    int a = cu[i], b = cu[i + 1];
    if (q_base >= a && q_base < b) { seg_start = a; seg_end = b; }
  }

  const int qrow = q_base + lr;
  const float* qp = Q + (size_t)qrow * TOK_STRIDE + head * HEAD_DIM;

  v16h qb[3];
#pragma unroll
  for (int c = 0; c < 3; ++c) {
    const int d0 = c * 32 + half * 16;
#pragma unroll
    for (int e = 0; e < 16; ++e) {
      const int d = d0 + e;
      qb[c][e] = (d < HEAD_DIM) ? (_Float16)(qp[d] * scale_log2e) : (_Float16)0.0f;
    }
  }

  v8f  o[5] = {};
  float m_run = NEG_BIG, l_run = 0.0f;

  for (int kvb = seg_start; kvb < seg_end; kvb += 32) {
    v8f st[2];
#pragma unroll
    for (int t = 0; t < 2; ++t) {
      v8f acc = {};
      int krow = kvb + 16 * t + lr;
      if (krow >= seg_end) krow = seg_end - 1;
      const float* kp = K + (size_t)krow * TOK_STRIDE + head * HEAD_DIM;
#pragma unroll
      for (int c = 0; c < 3; ++c) {
        v16h ka;
        const int d0 = c * 32 + half * 16;
#pragma unroll
        for (int e = 0; e < 8; ++e) {
          const int dA = d0 + e, dB = d0 + 8 + e;
          ka[e]     = (dA < HEAD_DIM) ? (_Float16)kp[dA] : (_Float16)0.0f;
          ka[8 + e] = (dB < HEAD_DIM) ? (_Float16)kp[dB] : (_Float16)0.0f;
        }
        acc = __builtin_amdgcn_wmma_f32_16x16x32_f16(false, ka, false, qb[c],
                                                     (short)0, acc, false, false);
      }
      st[t] = acc;
    }

    const int rem = seg_end - kvb;
    float mblk = NEG_BIG;
#pragma unroll
    for (int v = 0; v < 8; ++v) {
      if (v + 8 * half >= rem)      st[0][v] = NEG_BIG;
      if (16 + v + 8 * half >= rem) st[1][v] = NEG_BIG;
      mblk = fmaxf(mblk, fmaxf(st[0][v], st[1][v]));
    }
    mblk = fmaxf(mblk, __shfl_xor(mblk, 16, 32));
    const float mnew = fmaxf(m_run, mblk);
    const float corr = exp2f(m_run - mnew);
    m_run = mnew;

    float p0[8], p1[8], lblk = 0.0f;
#pragma unroll
    for (int v = 0; v < 8; ++v) {
      p0[v] = exp2f(st[0][v] - mnew);
      p1[v] = exp2f(st[1][v] - mnew);
      lblk += p0[v] + p1[v];
    }
    lblk += __shfl_xor(lblk, 16, 32);
    l_run = l_run * corr + lblk;
#pragma unroll
    for (int dt = 0; dt < 5; ++dt) o[dt] *= corr;

    v16h pb;
#pragma unroll
    for (int v = 0; v < 8; ++v) {
      const float s0 = __shfl_xor(p0[v], 16, 32);
      const float s1 = __shfl_xor(p1[v], 16, 32);
      pb[v]     = (_Float16)((half == 0) ? p0[v] : s1);
      pb[8 + v] = (_Float16)((half == 0) ? s0    : p1[v]);
    }

#pragma unroll
    for (int dt = 0; dt < 5; ++dt) {
      const int d = dt * 16 + lr;
      v16h va;
#pragma unroll
      for (int e = 0; e < 8; ++e) {
        int kv0 = kvb + 16 * half + e;
        int kv1 = kvb + 16 * half + 8 + e;
        if (kv0 >= seg_end) kv0 = seg_end - 1;
        if (kv1 >= seg_end) kv1 = seg_end - 1;
        va[e]     = (_Float16)V[(size_t)kv0 * TOK_STRIDE + head * HEAD_DIM + d];
        va[8 + e] = (_Float16)V[(size_t)kv1 * TOK_STRIDE + head * HEAD_DIM + d];
      }
      o[dt] = __builtin_amdgcn_wmma_f32_16x16x32_f16(false, va, false, pb,
                                                     (short)0, o[dt], false, false);
    }
  }

  const float linv = 1.0f / l_run;
  float* op = Out + (size_t)qrow * TOK_STRIDE + head * HEAD_DIM;
#pragma unroll
  for (int dt = 0; dt < 5; ++dt)
#pragma unroll
    for (int v = 0; v < 8; ++v)
      op[dt * 16 + v + 8 * half] = o[dt][v] * linv;
}

// ---------------- launcher ----------------
extern "C" void kernel_launch(void* const* d_in, const int* in_sizes, int n_in,
                              void* d_out, int out_size, void* d_ws, size_t ws_size,
                              hipStream_t stream) {
  const float* q  = (const float*)d_in[0];
  const float* k  = (const float*)d_in[1];
  const float* v  = (const float*)d_in[2];
  const int*   cu = (const int*)d_in[4];
  const int n_cu  = in_sizes[4];
  const int S     = in_sizes[0] / (NUM_HEADS * HEAD_DIM);

  const float scale_log2e = (1.0f / sqrtf((float)HEAD_DIM)) * 1.44269504088896340736f;

  const int Sv = (S + 7) & ~7;
  const size_t qk_bytes  = (size_t)NUM_HEADS * S * HEAD_DIM * sizeof(_Float16);
  const size_t vt_bytes  = (size_t)NUM_HEADS * HEAD_DIM * Sv * sizeof(_Float16);
  const size_t region_qk = (qk_bytes + 512 + 255) & ~(size_t)255;
  const size_t offQ = 0;
  const size_t offK = region_qk;
  const size_t offV = 2 * region_qk;
  const size_t need = offV + vt_bytes + 4096;

  const int qspan = 16 * QT * WAVES_PER_BLOCK;   // 128 query rows per block

  if (ws_size >= need && (S % qspan) == 0) {
    _Float16* Qf = (_Float16*)((char*)d_ws + offQ);
    _Float16* Kf = (_Float16*)((char*)d_ws + offK);
    _Float16* Vt = (_Float16*)((char*)d_ws + offV);

    const long pairs_qk = (long)NUM_HEADS * S * (HEAD_DIM / 2);
    cvt_qk_kernel<<<(int)((pairs_qk + 255) / 256), 256, 0, stream>>>(
        q, k, Qf, Kf, S, scale_log2e, pairs_qk);

    const long pairs_v = (long)NUM_HEADS * HEAD_DIM * (S / 2);
    cvt_vt_kernel<<<(int)((pairs_v + 255) / 256), 256, 0, stream>>>(
        v, Vt, S, Sv, pairs_v);

    dim3 grid(S / qspan, NUM_HEADS);
    fa_wmma_f16_kernel<<<grid, WAVES_PER_BLOCK * 32, 0, stream>>>(
        Qf, Kf, Vt, cu, n_cu, (float*)d_out, S, Sv);
  } else {
    const int qtiles = (S + 15) / 16;
    dim3 grid((qtiles + WAVES_PER_BLOCK - 1) / WAVES_PER_BLOCK, NUM_HEADS);
    fa_wmma_f32_kernel<<<grid, WAVES_PER_BLOCK * 32, 0, stream>>>(
        q, k, v, cu, n_cu, (float*)d_out, S, scale_log2e);
  }
}
